// SelfAttention_16741782520592
// MI455X (gfx1250) — matrix operand order repaired and checked by reading, not compiled, not run on hardware
//
#include <hip/hip_runtime.h>
#include <hip/hip_bf16.h>

typedef __attribute__((ext_vector_type(16))) _Float16 v16h;
typedef __attribute__((ext_vector_type(8)))  float    v8f;
typedef __attribute__((ext_vector_type(4)))  int      v4i;

#define BATCH 8
#define CCH   512
#define CQ    64
#define NPIX  4096

#if defined(__gfx1250__) && __has_builtin(__builtin_amdgcn_global_load_async_to_lds_b128) && __has_builtin(__builtin_amdgcn_s_wait_asynccnt)
#define USE_ASYNC 1
#else
#define USE_ASYNC 0
#endif

static __device__ __forceinline__ v8f wmma_f16(v16h a, v16h b, v8f c) {
  return __builtin_amdgcn_wmma_f32_16x16x32_f16(
      false, a, false, b, (short)0, c, false, false);
}

#if USE_ASYNC
static __device__ __forceinline__ void async_ld_b128(const void* g, void* l) {
  __builtin_amdgcn_global_load_async_to_lds_b128(
      (v4i*)g, (__attribute__((address_space(3))) v4i*)l, 0, 0);
}
#endif

// ---- fragment index helpers (16x16x32 f16 WMMA) ---------------------------
// A-fragment: lane = (row&15) + 16*half ; slot i<->K: half=(K>>3)&1,
//             slot=(K&7)+((K>>4)<<3)
// B-fragment: lane = (col&15) + 16*half ; slot i<->K same map as A (kaidx)
static __device__ __forceinline__ int kaidx(int i, int hi) { // hi = 0 or 8
  return ((i < 8) ? i : (i + 8)) + hi;                       // A-side K for slot i
}

// ---------------------------------------------------------------------------
// Projection: Z[o][n] = sum_c W[o][c]*x[b][c][n] + bias[o]; results written
// directly in packed WMMA-fragment order (f16):
//  kind 0 (K): B-fragments of energy GEMM   Kp[b][jt16][dt2][lane][slot]
//  kind 1 (Q): A-fragments of energy GEMM   Qp[b][mt16][kt2][lane][slot]
//  kind 2 (V): A-fragments of PV GEMM       Vp[b][jt32][ct16][lane][slot]
// One wave per 16x16 output tile, 4 waves/block.
// ---------------------------------------------------------------------------
__global__ void proj_wmma(const float* __restrict__ W,
                          const float* __restrict__ bias,
                          const float* __restrict__ x,
                          _Float16* __restrict__ out,
                          int O, int kind) {
  const int lane = threadIdx.x & 31;
  const int wave = threadIdx.x >> 5;
  const int lr   = lane & 15;
  const int hi   = (lane & 16) >> 1;  // 0 or 8

  const int tilesPerBatch = (O / 16) * (NPIX / 16);
  const int tile = blockIdx.x * 4 + wave;
  const int b  = tile / tilesPerBatch;
  const int rm = tile % tilesPerBatch;
  const int ot = rm / (NPIX / 16);
  const int nt = rm % (NPIX / 16);

  const int row = ot * 16 + lr;
  const int col = nt * 16 + lr;
  const float* xb = x + (size_t)b * CCH * NPIX;

  v8f acc = {};
  for (int k0 = 0; k0 < CCH; k0 += 32) {
    v16h af, bf;
#pragma unroll
    for (int i = 0; i < 16; ++i) {
      const int ka = k0 + kaidx(i, hi);
      const int kb = k0 + kaidx(i, hi);     // B-side K for slot i
      af[i] = (_Float16)W[(size_t)row * CCH + ka];
      bf[i] = (_Float16)xb[(size_t)kb * NPIX + col];
    }
    acc = wmma_f16(af, bf, acc);
  }

#pragma unroll
  for (int r = 0; r < 8; ++r) {
    const int o = ot * 16 + r + hi;         // output channel (D row)
    const int n = col;                      // spatial index (D col)
    const _Float16 v = (_Float16)(acc[r] + bias[o]);
    size_t idx;
    if (kind == 0) {                        // K: B-frag (d=o, j=n)
      const int jt = n >> 4, dt = o >> 5;
      const int ln = (n & 15) + 16 * ((o >> 3) & 1);
      const int sl = (o & 7) + (((o >> 4) & 1) << 3);
      idx = ((((size_t)b * 256 + jt) * 2 + dt) * 32 + ln) * 16 + sl;
    } else if (kind == 1) {                 // Q: A-frag (m=n, d=o)
      const int mt = n >> 4, kt = o >> 5, kk = o & 31;
      const int ln = (n & 15) + 16 * ((kk >> 3) & 1);
      const int sl = (kk & 7) + ((kk >> 4) << 3);
      idx = ((((size_t)b * 256 + mt) * 2 + kt) * 32 + ln) * 16 + sl;
    } else {                                // V: A-frag (c=o, j=n)
      const int jt = n >> 5, ct = o >> 4, kk = n & 31;
      const int ln = (o & 15) + 16 * (kk >> 4);
      const int sl = kk & 15;
      idx = ((((size_t)b * 128 + jt) * 32 + ct) * 32 + ln) * 16 + sl;
    }
    out[idx] = v;
  }
}

// ---------------------------------------------------------------------------
// Pass A: per-row softmax stats (rowmax/rowsum). One wave per 16-row query
// tile; fragments load as contiguous b128 pairs from the packed layouts.
// ---------------------------------------------------------------------------
__global__ void attn_stats(const _Float16* __restrict__ Qp,
                           const _Float16* __restrict__ Kp,
                           float* __restrict__ rowmax,
                           float* __restrict__ rowsum) {
  const int lane = threadIdx.x & 31;
  const int wave = threadIdx.x >> 5;
  const int lr   = lane & 15;
  const int hi   = (lane & 16) >> 1;

  const int gid = blockIdx.x * 4 + wave;     // 0..2047
  const int b   = gid >> 8;
  const int mt  = gid & 255;

  const v16h* qt = (const v16h*)(Qp + (((size_t)b * 256 + mt) * 2) * 512);
  const v16h qa0 = qt[lane];
  const v16h qa1 = qt[32 + lane];

  float rmax[8], rsum[8];
#pragma unroll
  for (int r = 0; r < 8; ++r) { rmax[r] = -__builtin_inff(); rsum[r] = 0.0f; }

  for (int jt = 0; jt < 256; ++jt) {
    const v16h* kt = (const v16h*)(Kp + (((size_t)b * 256 + jt) * 2) * 512);
    const v16h kb0 = kt[lane];
    const v16h kb1 = kt[32 + lane];
    v8f e = {};
    e = wmma_f16(qa0, kb0, e);
    e = wmma_f16(qa1, kb1, e);
#pragma unroll
    for (int r = 0; r < 8; ++r) {
      const float ev = e[r];
      const float mo = rmax[r];
      const float mn = fmaxf(mo, ev);
      rsum[r] = rsum[r] * __expf(mo - mn) + __expf(ev - mn);
      rmax[r] = mn;
    }
  }

#pragma unroll
  for (int mask = 1; mask < 16; mask <<= 1) {
#pragma unroll
    for (int r = 0; r < 8; ++r) {
      const float om = __shfl_xor(rmax[r], mask, 32);
      const float os = __shfl_xor(rsum[r], mask, 32);
      const float M  = fmaxf(rmax[r], om);
      rsum[r] = rsum[r] * __expf(rmax[r] - M) + os * __expf(om - M);
      rmax[r] = M;
    }
  }

  if (lr == 0) {
#pragma unroll
    for (int r = 0; r < 8; ++r) {
      const int m = mt * 16 + r + hi;
      rowmax[(size_t)b * NPIX + m] = rmax[r];
      rowsum[(size_t)b * NPIX + m] = rsum[r];
    }
  }
}

// ---------------------------------------------------------------------------
// Pass B: out[b][c][m] = gamma/rowsum[m] * sum_j exp(e[m][j]-rowmax[m])*V[c][j]
//                        + x[b][c][m]
// Block = 256 thr (8 waves) covers a 32-query m-tile x all 512 channels.
// Waves 0/1 recompute the two 16-row energy tiles (WMMA) and park P in LDS in
// B-fragment order; every wave runs D = V * P (A=V so stores are coalesced).
// V tiles are double-buffered through LDS with async loads when available.
// ---------------------------------------------------------------------------
__global__ void attn_out(const _Float16* __restrict__ Qp,
                         const _Float16* __restrict__ Kp,
                         const _Float16* __restrict__ Vp,
                         const float* __restrict__ rowmax,
                         const float* __restrict__ rowsum,
                         const float* __restrict__ x,
                         const float* __restrict__ gamma,
                         float* __restrict__ out) {
  const int tid   = threadIdx.x;
  const int lane  = tid & 31;
  const int wave  = tid >> 5;          // 0..7
  const int lr    = lane & 15;
  const int hi    = (lane & 16) >> 1;

  const int b   = blockIdx.x >> 7;     // 8 batches
  const int mb  = blockIdx.x & 127;    // 32-row m-tile
  const int m0  = mb * 32;

  __shared__ _Float16 Pt[2][512];      // P as B-fragments, one per 16-row half
#if USE_ASYNC
  __shared__ _Float16 Vbuf[2][32 * 512];  // 32 ct-tiles x (32 lanes x 16 halves)
#endif

  // waves 0/1: Q fragments + row maxima for their 16-row half
  v16h qa0 = {}, qa1 = {};
  float rmaxv[8];
  if (wave < 2) {
    const int mt = (m0 >> 4) + wave;
    const v16h* qt = (const v16h*)(Qp + (((size_t)b * 256 + mt) * 2) * 512);
    qa0 = qt[lane];
    qa1 = qt[32 + lane];
#pragma unroll
    for (int r = 0; r < 8; ++r)
      rmaxv[r] = rowmax[(size_t)b * NPIX + mt * 16 + r + hi];
  }

  v8f acc[2][4] = {{{}, {}, {}, {}}, {{}, {}, {}, {}}};

  const _Float16* vbase = Vp + (size_t)b * 128 * 32 * 512;  // [jt32][ct][ln][sl]

#if USE_ASYNC
  // prologue: stage step 0 into buffer 0 (one contiguous 32KB slab)
  {
    const _Float16* src = vbase + (size_t)tid * 64;
    _Float16* dst = &Vbuf[0][tid * 64];
#pragma unroll
    for (int k = 0; k < 8; ++k)
      async_ld_b128(src + k * 8, dst + k * 8);
  }
  int pb_sel = 0;
#endif

  for (int jt32 = 0; jt32 < 128; ++jt32) {
#if USE_ASYNC
    // prefetch next step (clamped) into the other buffer
    {
      const int nxt = (jt32 < 127) ? jt32 + 1 : 127;
      const _Float16* src = vbase + ((size_t)nxt * 32 * 512) + (size_t)tid * 64;
      _Float16* dst = &Vbuf[1 - pb_sel][tid * 64];
#pragma unroll
      for (int k = 0; k < 8; ++k)
        async_ld_b128(src + k * 8, dst + k * 8);
    }
#endif

    // waves 0/1: energy tile -> exp -> P (B-fragment order) into LDS
    if (wave < 2) {
#pragma unroll
      for (int jt = 0; jt < 2; ++jt) {
        const int jt16 = jt32 * 2 + jt;
        const v16h* kt = (const v16h*)(Kp + (((size_t)b * 256 + jt16) * 2) * 512);
        const v16h kb0 = kt[lane];
        const v16h kb1 = kt[32 + lane];
        v8f e = {};
        e = wmma_f16(qa0, kb0, e);
        e = wmma_f16(qa1, kb1, e);
#pragma unroll
        for (int r = 0; r < 8; ++r) {
          // element (m_local=r+hi, kk=jt*16+lr) -> Pt[(m&15)+16*jt][lr]
          Pt[wave][((r + hi) + 16 * jt) * 16 + lr] =
              (_Float16)__expf(e[r] - rmaxv[r]);
        }
      }
    }

#if USE_ASYNC
    __builtin_amdgcn_s_wait_asynccnt(8);   // current buffer's 8 loads done
#endif
    __syncthreads();

    // B-fragments of P for both 16-row halves (contiguous LDS reads)
    const v16h pfrag0 = ((const v16h*)Pt[0])[lane];
    const v16h pfrag1 = ((const v16h*)Pt[1])[lane];

#pragma unroll
    for (int t = 0; t < 4; ++t) {
      const int ct = wave * 4 + t;
#if USE_ASYNC
      const v16h va = ((const v16h*)Vbuf[pb_sel])[ct * 32 + lane];
#else
      const v16h va =
          ((const v16h*)vbase)[((size_t)jt32 * 32 + ct) * 32 + lane];
#endif
      acc[0][t] = wmma_f16(va, pfrag0, acc[0][t]);
      acc[1][t] = wmma_f16(va, pfrag1, acc[1][t]);
    }
    __syncthreads();
#if USE_ASYNC
    pb_sel ^= 1;
#endif
  }

  // epilogue: scale by gamma/rowsum and add residual (coalesced: col = m)
  const float g = gamma[0];
#pragma unroll
  for (int ms = 0; ms < 2; ++ms) {
    const int m = m0 + ms * 16 + lr;               // D column for this lane
    const float s = g / rowsum[(size_t)b * NPIX + m];
#pragma unroll
    for (int t = 0; t < 4; ++t) {
      const int ctile = wave * 4 + t;
#pragma unroll
      for (int r = 0; r < 8; ++r) {
        const int c = ctile * 16 + r + hi;         // D row = channel
        const size_t idx = ((size_t)b * CCH + c) * NPIX + m;
        out[idx] = s * acc[ms][t][r] + x[idx];
      }
    }
  }
}

// ---------------------------------------------------------------------------
extern "C" void kernel_launch(void* const* d_in, const int* in_sizes, int n_in,
                              void* d_out, int out_size, void* d_ws, size_t ws_size,
                              hipStream_t stream) {
  const float* x     = (const float*)d_in[0];
  const float* wq    = (const float*)d_in[1];
  const float* bq    = (const float*)d_in[2];
  const float* wk    = (const float*)d_in[3];
  const float* bk    = (const float*)d_in[4];
  const float* wv    = (const float*)d_in[5];
  const float* bv    = (const float*)d_in[6];
  const float* gamma = (const float*)d_in[7];
  float* out = (float*)d_out;

  char* ws = (char*)d_ws;
  const size_t qk_elems = (size_t)BATCH * NPIX * CQ;    // 2M halves each
  const size_t v_elems  = (size_t)BATCH * NPIX * CCH;   // 16M halves
  _Float16* Qp = (_Float16*)ws;
  _Float16* Kp = Qp + qk_elems;
  _Float16* Vp = Kp + qk_elems;
  float* rowmax = (float*)(Vp + v_elems);
  float* rowsum = rowmax + (size_t)BATCH * NPIX;

  // Projections (1 wave per 16x16 tile, 4 waves/block)
  proj_wmma<<<2048, 128, 0, stream>>>(wq, bq, x, Qp, CQ, 1);
  proj_wmma<<<2048, 128, 0, stream>>>(wk, bk, x, Kp, CQ, 0);
  proj_wmma<<<16384, 128, 0, stream>>>(wv, bv, x, Vp, CCH, 2);

  // Softmax stats: 2048 waves
  attn_stats<<<512, 128, 0, stream>>>(Qp, Kp, rowmax, rowsum);

  // Output: one block per (batch, 32-row m-tile)
  attn_out<<<1024, 256, 0, stream>>>(Qp, Kp, Vp, rowmax, rowsum, x, gamma, out);
}
